// DeformCrossAttention_7842610283318
// MI455X (gfx1250) — matrix operand order repaired and checked by reading, not compiled, not run on hardware
//
#include <hip/hip_runtime.h>
#include <hip/hip_bf16.h>
#include <math.h>

// ---------------------------------------------------------------------------
// CDNA5 (gfx1250) deformable cross attention.
// Conv stacks + QK^T + A*V on v_wmma_f32_16x16x32_f16.
// Conv B-tiles staged to LDS via TDM (tensor_load_to_lds) / async-to-LDS with
// double buffering; attention GEMMs are dense vector-load + WMMA loops.
// ---------------------------------------------------------------------------

typedef __attribute__((ext_vector_type(16))) _Float16 v16h;
typedef __attribute__((ext_vector_type(8)))  _Float16 v8h;
typedef __attribute__((ext_vector_type(8)))  float    v8f;
typedef __attribute__((ext_vector_type(4)))  unsigned u32x4;
typedef __attribute__((ext_vector_type(8)))  int      i32x8;
typedef __attribute__((ext_vector_type(4)))  int      i32x4;
typedef int v2i32 __attribute__((vector_size(8)));   // async b64 payload type

#ifdef __has_builtin
#  if __has_builtin(__builtin_amdgcn_global_load_async_to_lds_b64) && \
      __has_builtin(__builtin_amdgcn_s_wait_asynccnt)
#    define HAS_ASYNC 1
#  endif
#  if __has_builtin(__builtin_amdgcn_tensor_load_to_lds) && \
      __has_builtin(__builtin_amdgcn_s_wait_tensorcnt)
#    define HAS_TDM 1
#  endif
#endif
#ifndef HAS_ASYNC
#  define HAS_ASYNC 0
#endif
#ifndef HAS_TDM
#  define HAS_TDM 0
#endif
#if HAS_TDM && defined(__has_include)
#  if __has_include(<hip/amd_detail/amd_gfx1250_TDM.h>)
#    define TDM_ARGS6 1
#  endif
#endif
#ifndef TDM_ARGS6
#  define TDM_ARGS6 0
#endif

#define NHEAD   4
#define H_IN    256
#define W_IN    320
#define HMAX    320              // >= max Hp (ph<=57 -> Hp<=306)
#define WMAX    352              // >= max Wp (pw<=30 -> Wp<=348)
#define PH_     (HMAX + 4)
#define PW_     (WMAX + 4)
#define NTOKMAX ((HMAX / 8) * (WMAX / 8))   // 1760
#define DTOKMAX (16 * 57 * 30)              // 27360
#define PLANE   ((size_t)HMAX * WMAX)

// dims[8] per sample: ph, pw, nph, npw, Hp, Wp, ntok, Dtok

// -------------------------------- WMMA helpers -----------------------------

__device__ __forceinline__ v8f wmma16(const v16h& a, const v16h& b, v8f c) {
  return __builtin_amdgcn_wmma_f32_16x16x32_f16(false, a, false, b, (short)0, c,
                                                false, false);
}
__device__ __forceinline__ v16h combine16(v8h lo, v8h hi) {
  return __builtin_shufflevector(lo, hi, 0, 1, 2, 3, 4, 5, 6, 7, 8, 9, 10, 11,
                                 12, 13, 14, 15);
}
// byte offset of a __shared__ object within the LDS aperture
__device__ __forceinline__ unsigned lds_off(void* p) {
  return (unsigned)(unsigned long long)(__attribute__((address_space(3))) void*)p;
}

// ------------------------------- small utils -------------------------------

__global__ void k_zero_h16(_Float16* p, size_t n) {
  for (size_t i = (size_t)blockIdx.x * blockDim.x + threadIdx.x; i < n;
       i += (size_t)gridDim.x * blockDim.x)
    p[i] = (_Float16)0.f;
}
__global__ void k_zero_f32(float* p, size_t n) {
  for (size_t i = (size_t)blockIdx.x * blockDim.x + threadIdx.x; i < n;
       i += (size_t)gridDim.x * blockDim.x)
    p[i] = 0.f;
}

// reflect-pad H_IN x W_IN -> Hp x Wp, fp32 -> f16 into conv-halo buffer
__global__ void k_pad_reflect(const float* __restrict__ src, _Float16* dst,
                              int C, int choff, const int* __restrict__ dims) {
  const int Hp = dims[4], Wp = dims[5];
  size_t total = (size_t)C * HMAX * WMAX;
  for (size_t idx = (size_t)blockIdx.x * 256 + threadIdx.x; idx < total;
       idx += (size_t)gridDim.x * 256) {
    int x = (int)(idx % WMAX);
    size_t t = idx / WMAX;
    int y = (int)(t % HMAX);
    int c = (int)(t / HMAX);
    if (y >= Hp || x >= Wp) continue;
    int sy = (y < H_IN) ? y : (2 * H_IN - 2 - y);
    int sx = (x < W_IN) ? x : (2 * W_IN - 2 - x);
    dst[((size_t)(choff + c) * PH_ + (y + 2)) * PW_ + (x + 2)] =
        (_Float16)src[((size_t)c * H_IN + sy) * W_IN + sx];
  }
}

// ---------------------- implicit-GEMM 5x5 conv (WMMA) ----------------------
// grid: (3, HMAX, cout/16); block 256 = 8 waves; wave = 16 cout x 16 pixels.
// B tile (32 cin x 132 cols, one input row) staged into Raw[2] by TDM and/or
// async-to-LDS (double buffered), transposed in LDS, consumed by WMMA.
template <int CIN>
__global__ void __launch_bounds__(256)
k_conv5x5(const _Float16* __restrict__ xp, const float* __restrict__ wgt,
          const float* __restrict__ bias, float* __restrict__ out,
          const int* __restrict__ dims) {
  const int Hp = dims[4], Wp = dims[5];
  const int y = blockIdx.y;
  const int x0 = blockIdx.x * 128;
  if (y >= Hp || x0 >= Wp) return;
  const int coz = blockIdx.z;
  const int tid = threadIdx.x, lane = tid & 31, wv = tid >> 5;
  const int aoff = (lane < 16) ? 0 : 8;
  const int boff = (lane < 16) ? 0 : 8;

  __shared__ _Float16 Raw[2][32][132];  // row-major staging (double buffer)
  __shared__ _Float16 Bst[132][40];     // [col][cin32] transposed
  __shared__ _Float16 As[16][32];       // [cout16][cin32]

  constexpr int NC = CIN / 32;
  constexpr int T = 5 * NC;
  const size_t planeP = (size_t)PH_ * PW_;

  // does tile t use the TDM engine (when both engines are available)?
  auto tile_uses_tdm = [&](int t) -> bool {
#if HAS_TDM && HAS_ASYNC
    return (t & 1) == 0;
#elif HAS_TDM
    return true;
#else
    (void)t; return false;
#endif
  };

  auto issue_tile = [&](int t, int buf) {
    const int r = t / NC, ckb = (t - r * NC) * 32;
    const _Float16* gbase = xp + (size_t)ckb * planeP + (size_t)(y + r) * PW_ + x0;
#if HAS_TDM
    if (tile_uses_tdm(t)) {
      if (wv == 0) {
        unsigned laddr = lds_off(&Raw[buf][0][0]);
        unsigned long long ga = (unsigned long long)gbase;
        u32x4 g0 = {1u,                                  // count=1, user mode
                    laddr,                               // lds_addr
                    (unsigned)ga,                        // global_addr[31:0]
                    (unsigned)((ga >> 32) & 0x1FFFFFFull) | (2u << 30)};
        // group1: data_size=2B; tensor_dim0/1 = 1<<30; tile 132 x 32;
        // tensor_dim0_stride = channel plane stride (elements)
        i32x8 g1 = {0x00010000,
                    0,
                    0x4000,
                    0x4000 | (132 << 16),
                    32,
                    (int)((size_t)PH_ * PW_),
                    0,
                    0};
        i32x4 gz = {0, 0, 0, 0};
#if TDM_ARGS6
        i32x8 gz8 = {0, 0, 0, 0, 0, 0, 0, 0};
        __builtin_amdgcn_tensor_load_to_lds(g0, g1, gz, gz, gz8, 0);
#else
        __builtin_amdgcn_tensor_load_to_lds(g0, g1, gz, gz, 0);
#endif
      }
      return;
    }
#endif
#if HAS_ASYNC
    {
      // per-lane async b64 chunks: 32 rows x 33 chunks (4 halves each)
      for (int c = tid; c < 32 * 33; c += 256) {
        int ch = c / 33, j = c - ch * 33;
        const _Float16* g = gbase + (size_t)ch * planeP + j * 4;
        _Float16* l = &Raw[buf][ch][j * 4];
        __builtin_amdgcn_global_load_async_to_lds_b64(
            (__attribute__((address_space(1))) v2i32*)(unsigned long long)(const void*)g,
            (__attribute__((address_space(3))) v2i32*)(unsigned long long)lds_off(l),
            0, 0);
      }
      return;
    }
#else
    {
      // synchronous fallback staging
      for (int c = tid; c < 32 * 132; c += 256) {
        int ch = c / 132, col = c - ch * 132;
        int gx = x0 + col;
        _Float16 v = (_Float16)0.f;
        if (gx < PW_) v = xp[((size_t)(ckb + ch)) * planeP + (size_t)(y + r) * PW_ + gx];
        Raw[buf][ch][col] = v;
      }
    }
#endif
  };

  auto wait_tile = [&](int t) {
#if HAS_TDM
    if (tile_uses_tdm(t)) {
      if (wv == 0) __builtin_amdgcn_s_wait_tensorcnt(0);
      return;
    }
#endif
#if HAS_ASYNC
    __builtin_amdgcn_s_wait_asynccnt(0);
#endif
    (void)t;
  };

  v8f acc = {0.f, 0.f, 0.f, 0.f, 0.f, 0.f, 0.f, 0.f};

  issue_tile(0, 0);
  int cur = 0;
  for (int t = 0; t < T; ++t) {
    wait_tile(t);
    __syncthreads();                      // tile t resident; Bst reads done
    if (t + 1 < T) issue_tile(t + 1, cur ^ 1);
    // transpose Raw[cur] -> Bst (LDS->LDS, overlapped with next tile's DMA)
    for (int c = tid; c < 32 * 132; c += 256) {
      int ch = c / 132, col = c - ch * 132;
      Bst[col][ch] = Raw[cur][ch][col];
    }
    const int r = t / NC, ckb = (t - (t / NC) * NC) * 32;
    for (int s = 0; s < 5; ++s) {
      __syncthreads();
      for (int idx = tid; idx < 16 * 32; idx += 256) {
        int mr = idx >> 5, kk = idx & 31;
        As[mr][kk] = (_Float16)
            wgt[(((size_t)(coz * 16 + mr) * CIN + (ckb + kk)) * 5 + r) * 5 + s];
      }
      __syncthreads();
      const int m = lane & 15;
      const int ncol = wv * 16 + (lane & 15) + s;
      v8h alo = *(const v8h*)(&As[m][aoff]);
      v8h ahi = *(const v8h*)(&As[m][16 + aoff]);
      v8h blo = *(const v8h*)(&Bst[ncol][boff]);
      v8h bhi = *(const v8h*)(&Bst[ncol][boff + 16]);
      acc = wmma16(combine16(alo, ahi), combine16(blo, bhi), acc);
    }
    cur ^= 1;
  }
  #pragma unroll
  for (int rr = 0; rr < 8; ++rr) {
    int co = coz * 16 + rr + ((lane < 16) ? 0 : 8);
    int px = x0 + wv * 16 + (lane & 15);
    if (px < Wp)
      out[((size_t)co * HMAX + y) * WMAX + px] = acc[rr] + bias[co];
  }
}

// ------------------------ BatchNorm (per-channel) --------------------------

__global__ void __launch_bounds__(256)
k_bn_stats(const float* __restrict__ x, float* mean, float* rstd,
           const int* __restrict__ dims) {
  const int c = blockIdx.x;
  const int Hp = dims[4], Wp = dims[5];
  const int N = Hp * Wp;
  __shared__ float ra[256], rb[256];
  float s = 0.f, s2 = 0.f;
  for (int i = threadIdx.x; i < N; i += 256) {
    int yy = i / Wp, xx = i - yy * Wp;
    float v = x[((size_t)c * HMAX + yy) * WMAX + xx];
    s += v; s2 += v * v;
  }
  ra[threadIdx.x] = s; rb[threadIdx.x] = s2; __syncthreads();
  for (int k = 128; k > 0; k >>= 1) {
    if (threadIdx.x < k) { ra[threadIdx.x] += ra[threadIdx.x + k];
                           rb[threadIdx.x] += rb[threadIdx.x + k]; }
    __syncthreads();
  }
  if (threadIdx.x == 0) {
    float m = ra[0] / N;
    float var = rb[0] / N - m * m;
    mean[c] = m;
    rstd[c] = rsqrtf(var + 1e-5f);
  }
}

__global__ void __launch_bounds__(256)
k_bn_relu_pad(const float* __restrict__ x, const float* __restrict__ mean,
              const float* __restrict__ rstd, _Float16* dst, int C,
              int use_bn, int use_relu, const int* __restrict__ dims) {
  const int Hp = dims[4], Wp = dims[5];
  size_t total = (size_t)C * HMAX * WMAX;
  for (size_t idx = (size_t)blockIdx.x * 256 + threadIdx.x; idx < total;
       idx += (size_t)gridDim.x * 256) {
    int xx = (int)(idx % WMAX);
    size_t t = idx / WMAX;
    int yy = (int)(t % HMAX);
    int c = (int)(t / HMAX);
    if (yy >= Hp || xx >= Wp) continue;
    float v = x[((size_t)c * HMAX + yy) * WMAX + xx];
    if (use_bn) v = (v - mean[c]) * rstd[c];
    if (use_relu) v = fmaxf(v, 0.f);
    dst[((size_t)c * PH_ + (yy + 2)) * PW_ + (xx + 2)] = (_Float16)v;
  }
}

// ---------------------------- scale/ratio path -----------------------------

__global__ void __launch_bounds__(256)
k_srconv1(const float* __restrict__ vis, const float* __restrict__ ir,
          const float* __restrict__ w, const float* __restrict__ b,
          float* out) {
  const int OH = 128, OW = 160;
  int idx = blockIdx.x * 256 + threadIdx.x;
  if (idx >= 2 * 4 * OH * OW) return;
  int ox = idx % OW; int t = idx / OW;
  int oy = t % OH; t /= OH;
  int co = t % 4; int bb = t / 4;
  float acc = b[co];
  for (int ci = 0; ci < 128; ++ci) {
    const float* src = (ci < 64)
        ? vis + ((size_t)(bb * 64 + ci)) * H_IN * W_IN
        : ir + ((size_t)(bb * 64 + (ci - 64))) * H_IN * W_IN;
    for (int ky = 0; ky < 3; ++ky) {
      int iy = oy * 2 - 1 + ky; if ((unsigned)iy >= H_IN) continue;
      for (int kx = 0; kx < 3; ++kx) {
        int ix = ox * 2 - 1 + kx; if ((unsigned)ix >= W_IN) continue;
        acc += src[iy * W_IN + ix] * w[((co * 128 + ci) * 3 + ky) * 3 + kx];
      }
    }
  }
  out[idx] = acc;
}

__global__ void __launch_bounds__(256)
k_srconv2(const float* __restrict__ in, const float* __restrict__ w,
          const float* __restrict__ b, float* out) {
  const int OH = 128, OW = 160;
  int idx = blockIdx.x * 256 + threadIdx.x;
  if (idx >= 2 * 4 * OH * OW) return;
  int ox = idx % OW; int t = idx / OW;
  int oy = t % OH; t /= OH;
  int co = t % 4; int bb = t / 4;
  float acc = b[co];
  for (int ci = 0; ci < 4; ++ci) {
    const float* src = in + ((size_t)(bb * 4 + ci)) * OH * OW;
    for (int ky = 0; ky < 3; ++ky) {
      int iy = oy - 1 + ky; if ((unsigned)iy >= OH) continue;
      for (int kx = 0; kx < 3; ++kx) {
        int ix = ox - 1 + kx; if ((unsigned)ix >= OW) continue;
        acc += src[iy * OW + ix] * w[((co * 4 + ci) * 3 + ky) * 3 + kx];
      }
    }
  }
  out[idx] = acc;
}

__global__ void __launch_bounds__(256)
k_bn_relu_small(float* x, int Bn, int C, int HW) {
  const int c = blockIdx.x;
  __shared__ float ra[256], rb[256], mv[2];
  float s = 0.f, s2 = 0.f;
  for (int bb = 0; bb < Bn; ++bb)
    for (int i = threadIdx.x; i < HW; i += 256) {
      float v = x[((size_t)(bb * C + c)) * HW + i];
      s += v; s2 += v * v;
    }
  ra[threadIdx.x] = s; rb[threadIdx.x] = s2; __syncthreads();
  for (int k = 128; k > 0; k >>= 1) {
    if (threadIdx.x < k) { ra[threadIdx.x] += ra[threadIdx.x + k];
                           rb[threadIdx.x] += rb[threadIdx.x + k]; }
    __syncthreads();
  }
  if (threadIdx.x == 0) {
    float m = ra[0] / (Bn * HW);
    float var = rb[0] / (Bn * HW) - m * m;
    mv[0] = m; mv[1] = rsqrtf(var + 1e-5f);
  }
  __syncthreads();
  float m = mv[0], rs = mv[1];
  for (int bb = 0; bb < Bn; ++bb)
    for (int i = threadIdx.x; i < HW; i += 256) {
      size_t o = ((size_t)(bb * C + c)) * HW + i;
      x[o] = fmaxf((x[o] - m) * rs, 0.f);
    }
}

__global__ void __launch_bounds__(256)
k_sr_head(const float* __restrict__ x, const float* s1w, const float* s1b,
          const float* s2w, const float* s2b, const float* r1w,
          const float* r1b, const float* r2w, const float* r2b,
          int* dims_all) {
  const int bb = blockIdx.x;
  const int HW = 128 * 160;
  __shared__ float red[256];
  __shared__ float g[4];
  for (int c = 0; c < 4; ++c) {
    float s = 0.f;
    for (int i = threadIdx.x; i < HW; i += 256)
      s += x[((size_t)(bb * 4 + c)) * HW + i];
    red[threadIdx.x] = s; __syncthreads();
    for (int k = 128; k > 0; k >>= 1) {
      if (threadIdx.x < k) red[threadIdx.x] += red[threadIdx.x + k];
      __syncthreads();
    }
    if (threadIdx.x == 0) g[c] = red[0] / HW;
    __syncthreads();
  }
  if (threadIdx.x == 0) {
    auto mlp = [&](const float* w1, const float* b1, const float* w2,
                   const float* b2) {
      float acc = b2[0];
      for (int j = 0; j < 4; ++j) {
        float h = b1[j];
        for (int k = 0; k < 4; ++k) h += g[k] * w1[j * 4 + k];
        float ge = 0.5f * h * (1.f + erff(h * 0.70710678f));
        acc += ge * w2[j];
      }
      return tanhf(acc);
    };
    float scale = 0.9f * mlp(s1w, s1b, s2w, s2b) + 1.f;
    float ratio = 0.9f * mlp(r1w, r1b, r2w, r2b) + 1.f;
    int pw = (int)roundf(scale * 16.f); pw = pw < 8 ? 8 : (pw > 30 ? 30 : pw);
    int ph = (int)roundf(ratio * (float)pw);
    ph = ph < 8 ? 8 : (ph > 57 ? 57 : ph);
    int nph = (H_IN + ph - 1) / ph, npw = (W_IN + pw - 1) / pw;
    int* d = dims_all + bb * 8;
    d[0] = ph; d[1] = pw; d[2] = nph; d[3] = npw;
    d[4] = nph * ph; d[5] = npw * pw; d[6] = nph * npw; d[7] = 16 * ph * pw;
  }
}

// ------------------------------- tokens / LN -------------------------------

struct TokSrc { const float* T; int chb; int t; };

__device__ __forceinline__ TokSrc
tok_resolve(int kind, int h, int tt, int ntok, const float* vqk,
            const float* iqk, const float* gv) {
  TokSrc r;
  if (kind == 0)      { r.T = vqk; r.chb = h * 16;      r.t = tt; }
  else if (kind == 1) { r.T = vqk; r.chb = 64 + h * 16; r.t = tt; }
  else if (kind == 2) { r.T = iqk; r.chb = h * 16;      r.t = tt; }
  else if (kind == 3) { r.T = iqk; r.chb = 64 + h * 16; r.t = tt; }
  else {
    int cp = tt / ntok;
    r.T = gv; r.chb = ((kind == 4) ? 0 : 128) + (h * 2 + cp) * 16;
    r.t = tt - cp * ntok;
  }
  return r;
}
__device__ __forceinline__ const float*
tok_base(const TokSrc& ts, int ph, int pw, int npw) {
  int pi = ts.t / npw, pj = ts.t - pi * npw;
  return ts.T + (size_t)ts.chb * PLANE + (size_t)(pi * ph) * WMAX + pj * pw;
}
__device__ __forceinline__ int ln_idx(int kind, int h, int tt) {
  return (kind * NHEAD + h) * (2 * NTOKMAX) + tt;
}

__global__ void k_build_gtab(int* G, const int* __restrict__ dims) {
  const int ph = dims[0], pw = dims[1], Dtok = dims[7];
  int f = blockIdx.x * 256 + threadIdx.x;
  if (f >= Dtok) return;
  int pp = ph * pw;
  int cl = f / pp; int rem = f - cl * pp;
  int py = rem / pw; int px = rem - py * pw;
  G[f] = (int)(cl * PLANE) + py * WMAX + px;
}

__global__ void __launch_bounds__(256)
k_ln_stats(const float* __restrict__ vqk, const float* __restrict__ iqk,
           const float* __restrict__ gv, const int* __restrict__ G,
           float* lnm, float* lnr, const int* __restrict__ dims) {
  const int kind = blockIdx.z, h = blockIdx.y, tt = blockIdx.x;
  const int ntok = dims[6], Dtok = dims[7];
  const int ph = dims[0], pw = dims[1], npw = dims[3];
  const int ntk = (kind < 4) ? ntok : 2 * ntok;
  if (tt >= ntk) return;
  TokSrc ts = tok_resolve(kind, h, tt, ntok, vqk, iqk, gv);
  const float* base = tok_base(ts, ph, pw, npw);
  __shared__ float ra[256], rb[256];
  float s = 0.f, s2 = 0.f;
  for (int f = threadIdx.x; f < Dtok; f += 256) {
    float v = base[G[f]];
    s += v; s2 += v * v;
  }
  ra[threadIdx.x] = s; rb[threadIdx.x] = s2; __syncthreads();
  for (int k = 128; k > 0; k >>= 1) {
    if (threadIdx.x < k) { ra[threadIdx.x] += ra[threadIdx.x + k];
                           rb[threadIdx.x] += rb[threadIdx.x + k]; }
    __syncthreads();
  }
  if (threadIdx.x == 0) {
    float m = ra[0] / Dtok;
    float var = rb[0] / Dtok - m * m;
    int li = ln_idx(kind, h, tt);
    lnm[li] = m; lnr[li] = rsqrtf(var + 1e-5f);
  }
}

__global__ void __launch_bounds__(256)
k_tok_qk(const float* __restrict__ vqk, const float* __restrict__ iqk,
         const float* __restrict__ gv, const int* __restrict__ G,
         const float* __restrict__ lnm, const float* __restrict__ lnr,
         _Float16* __restrict__ qkm, size_t szqk,
         const int* __restrict__ dims) {
  const int kind = blockIdx.z, h = blockIdx.y, tt = blockIdx.x;
  const int ntok = dims[6], Dtok = dims[7];
  const int ph = dims[0], pw = dims[1], npw = dims[3];
  if (tt >= ntok) return;
  const int Dpad = (Dtok + 31) & ~31;
  TokSrc ts = tok_resolve(kind, h, tt, ntok, vqk, iqk, gv);
  const float* base = tok_base(ts, ph, pw, npw);
  const int li = ln_idx(kind, h, tt);
  const float m = lnm[li], rs = lnr[li];
  _Float16* drow = qkm + (size_t)(kind * NHEAD + h) * szqk + (size_t)tt * Dpad;
  for (int f = threadIdx.x; f < Dpad; f += 256) {
    float v = (f < Dtok) ? (base[G[f]] - m) * rs : 0.f;
    drow[f] = (_Float16)v;
  }
}

__global__ void __launch_bounds__(256)
k_tok_v(const float* __restrict__ vqk, const float* __restrict__ iqk,
        const float* __restrict__ gv, const int* __restrict__ G,
        const float* __restrict__ lnm, const float* __restrict__ lnr,
        _Float16* __restrict__ vtm, size_t szv,
        const int* __restrict__ dims) {
  const int pair = blockIdx.z, h = blockIdx.y, f = blockIdx.x;
  const int ntok = dims[6], Dtok = dims[7];
  const int ph = dims[0], pw = dims[1], npw = dims[3];
  if (f >= Dtok) return;
  const int kind = 4 + pair;
  const int M2 = 2 * ntok;
  const int Tpad = (M2 + 31) & ~31;
  const int g = G[f];
  _Float16* drow = vtm + (size_t)(pair * NHEAD + h) * szv + (size_t)f * Tpad;
  for (int tt = threadIdx.x; tt < Tpad; tt += 256) {
    float v = 0.f;
    if (tt < M2) {
      TokSrc ts = tok_resolve(kind, h, tt, ntok, vqk, iqk, gv);
      const float* base = tok_base(ts, ph, pw, npw);
      int li = ln_idx(kind, h, tt);
      v = (base[g] - lnm[li]) * lnr[li];
    }
    drow[tt] = (_Float16)v;
  }
}

// --------------------------- Q*K^T dense GEMM (WMMA) -----------------------
__global__ void __launch_bounds__(32)
k_qk_gemm(const _Float16* __restrict__ Qm, const _Float16* __restrict__ Km,
          size_t szqk, float* __restrict__ S, int colsel,
          const int* __restrict__ dims) {
  const int ntok = dims[6], Dtok = dims[7];
  const int Dpad = (Dtok + 31) & ~31;
  const int qt = blockIdx.x, kt = blockIdx.y, h = blockIdx.z;
  if (qt * 16 >= ntok || kt * 16 >= ntok) return;
  const int lane = threadIdx.x;
  int qm = qt * 16 + (lane & 15); if (qm >= ntok) qm = ntok - 1;
  int kn = kt * 16 + (lane & 15); if (kn >= ntok) kn = ntok - 1;
  const _Float16* qrow = Qm + (size_t)h * szqk + (size_t)qm * Dpad;
  const _Float16* krow = Km + (size_t)h * szqk + (size_t)kn * Dpad;
  const int aoff = (lane < 16) ? 0 : 8;
  const int boff = (lane < 16) ? 0 : 8;

  v8f acc = {0.f, 0.f, 0.f, 0.f, 0.f, 0.f, 0.f, 0.f};
  for (int kk = 0; kk < Dpad; kk += 32) {
    v8h alo = *(const v8h*)(qrow + kk + aoff);
    v8h ahi = *(const v8h*)(qrow + kk + 16 + aoff);
    v8h blo = *(const v8h*)(krow + kk + boff);
    v8h bhi = *(const v8h*)(krow + kk + boff + 16);
    acc = wmma16(combine16(alo, ahi), combine16(blo, bhi), acc);
  }
  const float scl = rsqrtf((float)Dtok);
  #pragma unroll
  for (int rr = 0; rr < 8; ++rr) {
    int qr = qt * 16 + rr + ((lane < 16) ? 0 : 8);
    int kc = kt * 16 + (lane & 15);
    if (qr < ntok && kc < ntok)
      S[((size_t)h * NTOKMAX + qr) * (2 * (size_t)NTOKMAX) +
        (size_t)colsel * ntok + kc] = acc[rr] * scl;
  }
}

// ------------------------------- row softmax -------------------------------
__global__ void __launch_bounds__(256)
k_softmax(float* __restrict__ S, const int* __restrict__ dims) {
  const int ntok = dims[6];
  const int row = blockIdx.x;
  if (row >= ntok) return;
  float* rp = S + ((size_t)(blockIdx.z * NHEAD + blockIdx.y) * NTOKMAX + row) *
                      (2 * (size_t)NTOKMAX);
  const int n2 = 2 * ntok;
  __shared__ float red[256];
  float mx = -1e30f;
  for (int i = threadIdx.x; i < n2; i += 256) mx = fmaxf(mx, rp[i]);
  red[threadIdx.x] = mx; __syncthreads();
  for (int k = 128; k > 0; k >>= 1) {
    if (threadIdx.x < k)
      red[threadIdx.x] = fmaxf(red[threadIdx.x], red[threadIdx.x + k]);
    __syncthreads();
  }
  mx = red[0]; __syncthreads();
  float sum = 0.f;
  for (int i = threadIdx.x; i < n2; i += 256) {
    float e = expf(rp[i] - mx);
    rp[i] = e; sum += e;
  }
  red[threadIdx.x] = sum; __syncthreads();
  for (int k = 128; k > 0; k >>= 1) {
    if (threadIdx.x < k) red[threadIdx.x] += red[threadIdx.x + k];
    __syncthreads();
  }
  float inv = 1.f / red[0];
  for (int i = threadIdx.x; i < n2; i += 256) rp[i] *= inv;
}

// -------------------- A*V dense GEMM (WMMA) + scatter ----------------------
__global__ void __launch_bounds__(32)
k_av_gemm(const float* __restrict__ Sp, const _Float16* __restrict__ Vt,
          size_t szv, float* __restrict__ outimg, int pair,
          const int* __restrict__ dims) {
  const int ntok = dims[6], Dtok = dims[7];
  const int ph = dims[0], pw = dims[1], npw = dims[3];
  const int qt = blockIdx.x, ft = blockIdx.y, h = blockIdx.z;
  if (qt * 16 >= ntok || ft * 16 >= Dtok) return;
  const int lane = threadIdx.x;
  const int M2 = 2 * ntok;
  const int Tpad = (M2 + 31) & ~31;
  int qm = qt * 16 + (lane & 15); if (qm >= ntok) qm = ntok - 1;
  int fn = ft * 16 + (lane & 15); if (fn >= Dtok) fn = Dtok - 1;
  const float* srow =
      Sp + ((size_t)h * NTOKMAX + qm) * (2 * (size_t)NTOKMAX);
  const _Float16* vrow = Vt + (size_t)h * szv + (size_t)fn * Tpad;
  const int aoff = (lane < 16) ? 0 : 8;
  const int boff = (lane < 16) ? 0 : 8;

  v8f acc = {0.f, 0.f, 0.f, 0.f, 0.f, 0.f, 0.f, 0.f};
  for (int kk = 0; kk < Tpad; kk += 32) {
    v8f af0 = *(const v8f*)(srow + kk + aoff);
    v8f af1 = *(const v8f*)(srow + kk + 16 + aoff);
    v16h a;
    #pragma unroll
    for (int i = 0; i < 8; ++i) {
      a[i] = (_Float16)af0[i];
      a[i + 8] = (_Float16)af1[i];
    }
    v8h blo = *(const v8h*)(vrow + kk + boff);
    v8h bhi = *(const v8h*)(vrow + kk + boff + 16);
    acc = wmma16(a, combine16(blo, bhi), acc);
  }
  const int pp = ph * pw;
  #pragma unroll
  for (int rr = 0; rr < 8; ++rr) {
    int q = qt * 16 + rr + ((lane < 16) ? 0 : 8);
    int f = ft * 16 + (lane & 15);
    if (q < ntok && f < Dtok) {
      int cl = f / pp; int rem = f - cl * pp;
      int py = rem / pw; int px = rem - py * pw;
      int pi = q / npw; int pj = q - pi * npw;
      int yy = pi * ph + py, xx = pj * pw + px;
      if (yy < H_IN && xx < W_IN)
        outimg[((size_t)(pair * 64 + h * 16 + cl) * H_IN + yy) * W_IN + xx] =
            acc[rr];
    }
  }
}

// =============================== host wiring ===============================

extern "C" void kernel_launch(void* const* d_in, const int* in_sizes, int n_in,
                              void* d_out, int out_size, void* d_ws,
                              size_t ws_size, hipStream_t stream) {
  (void)in_sizes; (void)n_in; (void)out_size; (void)ws_size;
  const float* vis = (const float*)d_in[0];
  const float* ir  = (const float*)d_in[1];
  const float* sr_c1_w = (const float*)d_in[2];
  const float* sr_c1_b = (const float*)d_in[3];
  const float* sr_c2_w = (const float*)d_in[4];
  const float* sr_c2_b = (const float*)d_in[5];
  const float* sr_s1_w = (const float*)d_in[6];
  const float* sr_s1_b = (const float*)d_in[7];
  const float* sr_s2_w = (const float*)d_in[8];
  const float* sr_s2_b = (const float*)d_in[9];
  const float* sr_r1_w = (const float*)d_in[10];
  const float* sr_r1_b = (const float*)d_in[11];
  const float* sr_r2_w = (const float*)d_in[12];
  const float* sr_r2_b = (const float*)d_in[13];
  const float* stk_w[3][4]; const float* stk_b[3][4];
  for (int s = 0; s < 3; ++s)
    for (int j = 0; j < 4; ++j) {
      stk_w[s][j] = (const float*)d_in[14 + s * 8 + j * 2];
      stk_b[s][j] = (const float*)d_in[14 + s * 8 + j * 2 + 1];
    }
  static const int cin_l[3][4]  = {{64, 64, 128, 128}, {64, 128, 128, 128},
                                   {128, 256, 256, 256}};
  static const int cout_l[3][4] = {{64, 128, 128, 128}, {128, 128, 128, 128},
                                   {256, 256, 256, 256}};

  char* ws = (char*)d_ws; size_t off = 0;
  auto alloc = [&](size_t bytes) {
    void* p = ws + off;
    off += (bytes + 255) & ~(size_t)255;
    return p;
  };
  const size_t PADPLANE = (size_t)PH_ * PW_;
  const size_t SZQK = (size_t)16 * HMAX * WMAX + 32 * NTOKMAX;
  const size_t SZV  = (size_t)32 * HMAX * WMAX + 32 * (DTOKMAX + 32);
  _Float16* padA   = (_Float16*)alloc(64  * PADPLANE * 2);
  _Float16* padB   = (_Float16*)alloc(64  * PADPLANE * 2);
  _Float16* padCat = (_Float16*)alloc(128 * PADPLANE * 2);
  _Float16* padC   = (_Float16*)alloc(256 * PADPLANE * 2);
  _Float16* padD   = (_Float16*)alloc(256 * PADPLANE * 2);
  float* cvA   = (float*)alloc(256 * PLANE * 4);
  float* cvB   = (float*)alloc(256 * PLANE * 4);
  float* visqk = (float*)alloc(128 * PLANE * 4);
  float* irqk  = (float*)alloc(128 * PLANE * 4);
  float* gvo   = (float*)alloc(256 * PLANE * 4);
  float* bnm   = (float*)alloc(256 * 4);
  float* bnr   = (float*)alloc(256 * 4);
  float* S     = (float*)alloc((size_t)2 * NHEAD * NTOKMAX * 2 * NTOKMAX * 4);
  float* lnm   = (float*)alloc((size_t)6 * NHEAD * 2 * NTOKMAX * 4);
  float* lnr   = (float*)alloc((size_t)6 * NHEAD * 2 * NTOKMAX * 4);
  float* sr1   = (float*)alloc((size_t)2 * 4 * 128 * 160 * 4);
  float* sr2   = (float*)alloc((size_t)2 * 4 * 128 * 160 * 4);
  int*   gtab  = (int*)alloc((size_t)DTOKMAX * 4);
  _Float16* qkm = (_Float16*)alloc((size_t)16 * SZQK * 2);
  _Float16* vtm = (_Float16*)alloc((size_t)8 * SZV * 2);
  int*   dims  = (int*)alloc(2 * 8 * sizeof(int));

  {
    int tot = 2 * 4 * 128 * 160;
    k_srconv1<<<(tot + 255) / 256, 256, 0, stream>>>(vis, ir, sr_c1_w, sr_c1_b,
                                                     sr1);
    k_bn_relu_small<<<4, 256, 0, stream>>>(sr1, 2, 4, 128 * 160);
    k_srconv2<<<(tot + 255) / 256, 256, 0, stream>>>(sr1, sr_c2_w, sr_c2_b,
                                                     sr2);
    k_bn_relu_small<<<4, 256, 0, stream>>>(sr2, 2, 4, 128 * 160);
    k_sr_head<<<2, 256, 0, stream>>>(sr2, sr_s1_w, sr_s1_b, sr_s2_w, sr_s2_b,
                                     sr_r1_w, sr_r1_b, sr_r2_w, sr_r2_b, dims);
  }

  auto run_conv = [&](int cin, const _Float16* in, const float* w,
                      const float* b, float* out, int cout, const int* dm) {
    dim3 grid(3, HMAX, cout / 16);
    if (cin == 64)
      k_conv5x5<64><<<grid, 256, 0, stream>>>(in, w, b, out, dm);
    else if (cin == 128)
      k_conv5x5<128><<<grid, 256, 0, stream>>>(in, w, b, out, dm);
    else
      k_conv5x5<256><<<grid, 256, 0, stream>>>(in, w, b, out, dm);
  };
  auto run_pad = [&](const float* x, int C, int use_bn, int use_relu,
                     _Float16* dst, const int* dm) {
    if (use_bn) k_bn_stats<<<C, 256, 0, stream>>>(x, bnm, bnr, dm);
    size_t tot = (size_t)C * HMAX * WMAX;
    k_bn_relu_pad<<<(unsigned)((tot + 255) / 256), 256, 0, stream>>>(
        x, bnm, bnr, dst, C, use_bn, use_relu, dm);
  };

  for (int i = 0; i < 2; ++i) {
    const int* dm = dims + i * 8;
    const float* visS = vis + (size_t)i * 64 * H_IN * W_IN;
    const float* irS  = ir  + (size_t)i * 64 * H_IN * W_IN;
    float* outS = (float*)d_out + (size_t)i * 128 * H_IN * W_IN;

    k_zero_h16<<<4096, 256, 0, stream>>>(padA, 64 * PADPLANE);
    k_zero_h16<<<4096, 256, 0, stream>>>(padB, 64 * PADPLANE);
    k_zero_h16<<<4096, 256, 0, stream>>>(padCat, 128 * PADPLANE);
    k_zero_h16<<<8192, 256, 0, stream>>>(padC, 256 * PADPLANE);
    k_zero_h16<<<8192, 256, 0, stream>>>(padD, 256 * PADPLANE);

    unsigned g64 = (unsigned)(((size_t)64 * HMAX * WMAX + 255) / 256);
    k_pad_reflect<<<g64, 256, 0, stream>>>(visS, padA, 64, 0, dm);
    k_pad_reflect<<<g64, 256, 0, stream>>>(irS, padB, 64, 0, dm);
    k_pad_reflect<<<g64, 256, 0, stream>>>(visS, padCat, 64, 0, dm);
    k_pad_reflect<<<g64, 256, 0, stream>>>(irS, padCat, 64, 64, dm);

    const _Float16* in0[3] = {padA, padB, padCat};
    float* fin[3] = {visqk, irqk, gvo};
    for (int s = 0; s < 3; ++s) {
      run_conv(cin_l[s][0], in0[s], stk_w[s][0], stk_b[s][0], cvA,
               cout_l[s][0], dm);
      run_pad(cvA, cout_l[s][0], 1, 1, padC, dm);
      run_conv(cin_l[s][1], padC, stk_w[s][1], stk_b[s][1], cvB, cout_l[s][1],
               dm);
      run_pad(cvB, cout_l[s][1], 1, 1, padD, dm);
      run_conv(cin_l[s][2], padD, stk_w[s][2], stk_b[s][2], cvA, cout_l[s][2],
               dm);
      run_pad(cvA, cout_l[s][2], 0, 1, padC, dm);
      run_conv(cin_l[s][3], padC, stk_w[s][3], stk_b[s][3], fin[s],
               cout_l[s][3], dm);
    }

    k_build_gtab<<<(DTOKMAX + 255) / 256, 256, 0, stream>>>(gtab, dm);
    {
      dim3 grid(2 * NTOKMAX, NHEAD, 6);
      k_ln_stats<<<grid, 256, 0, stream>>>(visqk, irqk, gvo, gtab, lnm, lnr,
                                           dm);
    }
    {
      dim3 grid(NTOKMAX, NHEAD, 4);
      k_tok_qk<<<grid, 256, 0, stream>>>(visqk, irqk, gvo, gtab, lnm, lnr,
                                         qkm, SZQK, dm);
    }
    {
      dim3 grid(DTOKMAX, NHEAD, 2);
      k_tok_v<<<grid, 256, 0, stream>>>(visqk, irqk, gvo, gtab, lnm, lnr,
                                        vtm, SZV, dm);
    }

    const size_t SPAIR = (size_t)NHEAD * NTOKMAX * 2 * NTOKMAX;
    k_zero_f32<<<16384, 256, 0, stream>>>(S, 2 * SPAIR);
    const _Float16* QKM[4] = {qkm + 0 * NHEAD * SZQK, qkm + 1 * NHEAD * SZQK,
                              qkm + 2 * NHEAD * SZQK, qkm + 3 * NHEAD * SZQK};
    {
      dim3 grid(NTOKMAX / 16, NTOKMAX / 16, NHEAD);
      k_qk_gemm<<<grid, 32, 0, stream>>>(QKM[0], QKM[1], SZQK, S, 0, dm);
      k_qk_gemm<<<grid, 32, 0, stream>>>(QKM[0], QKM[3], SZQK, S, 1, dm);
      k_qk_gemm<<<grid, 32, 0, stream>>>(QKM[2], QKM[3], SZQK, S + SPAIR, 0,
                                         dm);
      k_qk_gemm<<<grid, 32, 0, stream>>>(QKM[2], QKM[1], SZQK, S + SPAIR, 1,
                                         dm);
    }
    {
      dim3 grid(NTOKMAX, NHEAD, 2);
      k_softmax<<<grid, 256, 0, stream>>>(S, dm);
    }
    {
      dim3 grid(NTOKMAX / 16, (DTOKMAX + 15) / 16, NHEAD);
      k_av_gemm<<<grid, 32, 0, stream>>>(S, vtm + 0 * NHEAD * SZV, SZV, outS,
                                         0, dm);
      k_av_gemm<<<grid, 32, 0, stream>>>(S + SPAIR, vtm + 1 * NHEAD * SZV,
                                         SZV, outS, 1, dm);
    }
  }
}
